// EGNN_5995774345800
// MI455X (gfx1250) — matrix operand order repaired and checked by reading, not compiled, not run on hardware
//
#include <hip/hip_runtime.h>
#include <hip/hip_bf16.h>
#include <math.h>

typedef __attribute__((ext_vector_type(16))) _Float16 v16h;
typedef __attribute__((ext_vector_type(8)))  float    v8f;

#define NNODE 512
#define EROW  176   // padded LDS row stride (halves): 88 dwords -> conflict-free b128 reads

// ws layout (element offsets in _Float16 unless noted)
#define WS_W1EB      65536        // 46080 halves  : w1e B-fragments  (18 ntiles x 5 ksteps x 512)
#define WS_W2ET      111616       // 4608  halves  : w2e^T A-fragments (9 ksteps x 512)
#define WS_B1E_BYTE  232448       // 288 floats    : padded b1e

// ---------------------------------------------------------------------------
// Prep: f16 conversion + WMMA fragment swizzling of the edge-MLP weights.
// Fragment packing convention (16-bit, 16x16x32):
//   lane L, halves e=0..15 (v = e/2, pair = e%2, lh = L/16):
//   K_local = (v<4) ? lh*8 + 2v+pair : 16 + lh*8 + 2(v-4)+pair
//   A-frag: row M = L%16 ;  B-frag: col N = L%16
// ---------------------------------------------------------------------------
__global__ void egnn_prep(const float* __restrict__ feats,
                          const float* __restrict__ w1e,
                          const float* __restrict__ b1e,
                          const float* __restrict__ w2e,
                          _Float16* __restrict__ feats_h,
                          _Float16* __restrict__ w1eB,
                          _Float16* __restrict__ w2eT,
                          float* __restrict__ b1e_pad) {
  int t = blockIdx.x * 256 + threadIdx.x;
  if (t < 65536) feats_h[t] = (_Float16)feats[t];

  int u = t - 65536;                     // w1e B-fragments
  if (u >= 0 && u < 46080) {
    int tile = u >> 9, r = u & 511;
    int lane = r >> 4, e = r & 15;
    int nt = tile / 5, ks = tile % 5;
    int v = e >> 1, pair = e & 1, lh = lane >> 4;
    int kl = (v < 4) ? (lh * 8 + v * 2 + pair) : (16 + lh * 8 + (v - 4) * 2 + pair);
    int K = ks * 32 + kl;
    int n = nt * 16 + (lane & 15);
    float val = (K < 137 && n < 274) ? w1e[K * 274 + n] : 0.f;
    w1eB[u] = (_Float16)val;
  }
  int w = t - 65536 - 46080;             // w2e^T A-fragments
  if (w >= 0 && w < 4608) {
    int tile = w >> 9, r = w & 511;
    int lane = r >> 4, e = r & 15;
    int v = e >> 1, pair = e & 1, lh = lane >> 4;
    int kl = (v < 4) ? (lh * 8 + v * 2 + pair) : (16 + lh * 8 + (v - 4) * 2 + pair);
    int K = tile * 32 + kl;
    int n2 = lane & 15;
    float val = (K < 274) ? w2e[K * 16 + n2] : 0.f;
    w2eT[w] = (_Float16)val;
  }
  int x = t - 65536 - 46080 - 4608;      // padded b1e
  if (x >= 0 && x < 288) b1e_pad[x] = (x < 274) ? b1e[x] : 0.f;
}

__device__ inline v16h ld_afrag_lds(const _Float16* p16 /*16B aligned*/) {
  union { uint4 u[2]; v16h v; } tmp;
  tmp.u[0] = *(const uint4*)(p16);        // halves 0..7   (v0..v3)
  tmp.u[1] = *(const uint4*)(p16 + 16);   // halves 16..23 (v4..v7)
  return tmp.v;
}

// ---------------------------------------------------------------------------
// Main: one block per (b,i) node row; 8 waves x 4 chunks of 128 edges.
// ---------------------------------------------------------------------------
__global__ __launch_bounds__(256) void egnn_main(
    const float* __restrict__ feats, const float* __restrict__ coors,
    const float* __restrict__ b2e,
    const float* __restrict__ w1c, const float* __restrict__ b1c,
    const float* __restrict__ w2c, const float* __restrict__ b2c,
    const float* __restrict__ w1h, const float* __restrict__ b1h,
    const float* __restrict__ w2h, const float* __restrict__ b2h,
    const _Float16* __restrict__ feats_h, const _Float16* __restrict__ w1eB,
    const _Float16* __restrict__ w2eT, const float* __restrict__ b1e_pad,
    float* __restrict__ out) {

  __shared__ __align__(16) _Float16 lds_edge[128 * EROW];  // 45056 B
  __shared__ __align__(16) _Float16 m_all[NNODE * 16];     // 16384 B
  __shared__ float h3[128];
  __shared__ float mi_acc[16];
  __shared__ float coor_acc[4];

  const int bi = blockIdx.x;             // 0..1023
  const int b = bi >> 9, i = bi & 511;
  const int t = threadIdx.x;
  const int lane = t & 31, wave = t >> 5;
  const int lh = lane >> 4, lr = lane & 15;

  if (t < 16) mi_acc[t] = 0.f;
  if (t < 4)  coor_acc[t] = 0.f;

  const float cix = coors[(b * NNODE + i) * 3 + 0];
  const float ciy = coors[(b * NNODE + i) * 3 + 1];
  const float ciz = coors[(b * NNODE + i) * 3 + 2];

  float b2e_r[8];
#pragma unroll
  for (int v = 0; v < 8; ++v) b2e_r[v] = b2e[lh * 8 + v];

  for (int chunk = 0; chunk < 4; ++chunk) {
    __syncthreads();
    // ---- stage 128 edge-input rows (f16) into LDS (vectorized) ----
    {
      int e = t & 127, seg = t >> 7;
      int j = chunk * 128 + e;
      _Float16* row = &lds_edge[e * EROW];
      const uint4* fi4 = (const uint4*)&feats_h[(b * NNODE + i) * 64];
      const uint4* fj4 = (const uint4*)&feats_h[(b * NNODE + j) * 64];
      if (seg == 0) {
        // row[0..63] = feats_i ; row[64..79] = feats_j[0..15]
#pragma unroll
        for (int q = 0; q < 8; ++q) ((uint4*)row)[q] = fi4[q];
        ((uint4*)row)[8] = fj4[0];
        ((uint4*)row)[9] = fj4[1];
      } else {
        // row[80..127] = feats_j[16..63]
#pragma unroll
        for (int q = 0; q < 6; ++q) ((uint4*)row)[10 + q] = fj4[2 + q];
        float rx = cix - coors[(b * NNODE + j) * 3 + 0];
        float ry = ciy - coors[(b * NNODE + j) * 3 + 1];
        float rz = ciz - coors[(b * NNODE + j) * 3 + 2];
        float sq = rx * rx + ry * ry + rz * rz;
        float d = sq > 0.f ? sqrtf(sq) : 0.f;
        union { uint4 u[2]; _Float16 h[16]; } tail;  // row[128..143]
#pragma unroll
        for (int s = 0; s < 4; ++s) {
          float xs = d * (1.0f / (float)(1 << s));
          tail.h[s]     = (_Float16)sinf(xs);
          tail.h[4 + s] = (_Float16)cosf(xs);
        }
        tail.h[8] = (_Float16)d;
#pragma unroll
        for (int k = 9; k < 16; ++k) tail.h[k] = (_Float16)0.f;
        ((uint4*)row)[16] = tail.u[0];
        ((uint4*)row)[17] = tail.u[1];
        uint4 z = {0, 0, 0, 0};               // row[144..159] = 0
        ((uint4*)row)[18] = z;
        ((uint4*)row)[19] = z;
      }
    }
    __syncthreads();

    // ---- GEMM1 (edge @ w1e, ReLU) fused with GEMM2 (m^T = w2e^T x h^T) ----
    const int row0 = wave * 16 + lr;     // edge row this lane owns in A-frags
    v16h a[5];
#pragma unroll
    for (int ks = 0; ks < 5; ++ks)
      a[ks] = ld_afrag_lds(&lds_edge[row0 * EROW + ks * 32 + lh * 8]);

    v8f macc = {};
    v16h hh = {};
#pragma unroll 2   // keep (nt&1) static but bound register pressure (no spills)
    for (int nt = 0; nt < 18; ++nt) {
      v8f c = {};
#pragma unroll
      for (int ks = 0; ks < 5; ++ks) {
        v16h bfr = *(const v16h*)&w1eB[((nt * 5 + ks) * 32 + lane) * 16];
        c = __builtin_amdgcn_wmma_f32_16x16x32_f16(
                false, bfr, false, a[ks], (short)0, c, false, false);
      }
      const float* bias = &b1e_pad[nt * 16 + lh * 8];
      const int ho = (nt & 1) * 8;
#pragma unroll
      for (int v = 0; v < 8; ++v) {
        float hv = c[v] + bias[v];
        hh[ho + v] = (_Float16)(hv > 0.f ? hv : 0.f);
      }
      if (nt & 1) {  // two h N-tiles complete -> one 32-K step of GEMM2
        v16h wA = *(const v16h*)&w2eT[((nt >> 1) * 32 + lane) * 16];
        macc = __builtin_amdgcn_wmma_f32_16x16x32_f16(
                   false, wA, false, hh, (short)0, macc, false, false);
      }
    }
    // m^T C-frag: lane holds edge (lr), dims n2 = lh*8 + v  -> store to LDS
    {
      int jj = chunk * 128 + wave * 16 + lr;
      union { uint4 u; _Float16 hv[8]; } pk;
#pragma unroll
      for (int v = 0; v < 8; ++v) pk.hv[v] = (_Float16)(macc[v] + b2e_r[v]);
      *(uint4*)&m_all[jj * 16 + lh * 8] = pk.u;
    }
  }
  __syncthreads();

  // ---- phase 2a: coor MLP per edge + coors_out reduction ----
  {
    float cacc0 = 0.f, cacc1 = 0.f, cacc2 = 0.f;
#pragma unroll
    for (int rep = 0; rep < 2; ++rep) {
      int j = t + rep * 256;
      float m[16];
#pragma unroll
      for (int k = 0; k < 16; ++k) m[k] = (float)m_all[j * 16 + k];
      float cw = b2c[0];
      for (int o = 0; o < 64; ++o) {
        float acc = b1c[o];
#pragma unroll
        for (int k = 0; k < 16; ++k) acc += m[k] * w1c[k * 64 + o];
        cw += (acc > 0.f ? acc : 0.f) * w2c[o];
      }
      float rx = cix - coors[(b * NNODE + j) * 3 + 0];
      float ry = ciy - coors[(b * NNODE + j) * 3 + 1];
      float rz = ciz - coors[(b * NNODE + j) * 3 + 2];
      cacc0 += cw * rx; cacc1 += cw * ry; cacc2 += cw * rz;
    }
    atomicAdd(&coor_acc[0], cacc0);
    atomicAdd(&coor_acc[1], cacc1);
    atomicAdd(&coor_acc[2], cacc2);
  }
  // ---- phase 2b: m_i = sum_j m_ij ----
  {
    int k = t & 15, js = t >> 4;
    float s = 0.f;
    for (int j = js; j < NNODE; j += 16) s += (float)m_all[j * 16 + k];
    atomicAdd(&mi_acc[k], s);
  }
  __syncthreads();

  // ---- phase 2c: node MLP (80 -> 128 relu -> 64) ----
  if (t < 128) {
    float acc = b1h[t];
    const float* fi = &feats[(b * NNODE + i) * 64];
    for (int k = 0; k < 64; ++k) acc += fi[k] * w1h[k * 128 + t];
#pragma unroll
    for (int k = 0; k < 16; ++k) acc += mi_acc[k] * w1h[(64 + k) * 128 + t];
    h3[t] = acc > 0.f ? acc : 0.f;
  }
  __syncthreads();
  if (t < 64) {
    float acc = b2h[t];
    for (int o = 0; o < 128; ++o) acc += h3[o] * w2h[o * 64 + t];
    out[(b * NNODE + i) * 64 + t] = acc;
  }
  if (t < 3) out[65536 + (b * NNODE + i) * 3 + t] = coor_acc[t];
}

extern "C" void kernel_launch(void* const* d_in, const int* in_sizes, int n_in,
                              void* d_out, int out_size, void* d_ws, size_t ws_size,
                              hipStream_t stream) {
  const float* feats = (const float*)d_in[0];
  const float* coors = (const float*)d_in[1];
  const float* w1e   = (const float*)d_in[2];
  const float* b1e   = (const float*)d_in[3];
  const float* w2e   = (const float*)d_in[4];
  const float* b2e   = (const float*)d_in[5];
  const float* w1c   = (const float*)d_in[6];
  const float* b1c   = (const float*)d_in[7];
  const float* w2c   = (const float*)d_in[8];
  const float* b2c   = (const float*)d_in[9];
  const float* w1h   = (const float*)d_in[10];
  const float* b1h   = (const float*)d_in[11];
  const float* w2h   = (const float*)d_in[12];
  const float* b2h   = (const float*)d_in[13];

  _Float16* feats_h = (_Float16*)d_ws;
  _Float16* w1eB    = feats_h + WS_W1EB;
  _Float16* w2eT    = feats_h + WS_W2ET;
  float*    b1e_pad = (float*)((char*)d_ws + WS_B1E_BYTE);

  const int totalPrep = 65536 + 46080 + 4608 + 288;
  egnn_prep<<<(totalPrep + 255) / 256, 256, 0, stream>>>(
      feats, w1e, b1e, w2e, feats_h, w1eB, w2eT, b1e_pad);

  egnn_main<<<1024, 256, 0, stream>>>(
      feats, coors, b2e, w1c, b1c, w2c, b2c, w1h, b1h, w2h, b2h,
      feats_h, w1eB, w2eT, b1e_pad, (float*)d_out);
}
